// DCNv2_Pure_88270167867850
// MI455X (gfx1250) — matrix operand order repaired and checked by reading, not compiled, not run on hardware
//
#include <hip/hip_runtime.h>
#include <hip/hip_bf16.h>
#include <math.h>

#define B_   8
#define C_   64
#define CO_  64
#define H_   128
#define W_   128
#define HW_  (H_*W_)
#define KK_  9
#define CK_  (C_*KK_)      // 576
#define LDA_ 584           // padded LDS row length in halfs (breaks bank periodicity)
#define MTILE 128          // spatial pixels per workgroup (8 waves x 16 rows)

typedef __attribute__((ext_vector_type(16))) __bf16 v16bf;
typedef __attribute__((ext_vector_type(8)))  float  v8f;

union FragBF { uint4 u[2]; v16bf v; };

static __device__ __forceinline__ unsigned short f2bf(float f) {
    unsigned u = __float_as_uint(f);
    u += 0x7FFFu + ((u >> 16) & 1u);           // round-to-nearest-even
    return (unsigned short)(u >> 16);
}
static __device__ __forceinline__ int iclamp(int v, int lo, int hi) {
    return v < lo ? lo : (v > hi ? hi : v);
}

// ---------------------------------------------------------------------------
// Kernel 0: convert main conv weights to bf16 once. Layout [o][c*9+kk] is the
// natural flattening of (Cout, C, 3, 3), so this is a pure elementwise cast.
// ---------------------------------------------------------------------------
__global__ void dcn_prep_w(const float* __restrict__ w,
                           unsigned short* __restrict__ wbf) {
    int i = blockIdx.x * blockDim.x + threadIdx.x;
    if (i < CO_ * CK_) wbf[i] = f2bf(w[i]);
}

// ---------------------------------------------------------------------------
// Kernel 1: offset/mask 3x3 conv (64 -> 27 ch), fp32, LDS-tiled.
// Writes per-pixel [off_x(9) | off_y(9) | sigmoid(mask)(9)] to ws.
// ---------------------------------------------------------------------------
__global__ __launch_bounds__(256) void dcn_offsets(const float* __restrict__ x,
                                                   const float* __restrict__ ow,
                                                   const float* __restrict__ obv,
                                                   float* __restrict__ om) {
    __shared__ float xt[16 * 18 * 20];   // 16 ch x 18 rows x 20 (padded) cols
    __shared__ float wl[27 * 16 * 9];    // 27 out x 16 ch x 9 taps
    const int b = blockIdx.z, h0 = blockIdx.y * 16, w0 = blockIdx.x * 16;
    const int tid = threadIdx.y * 16 + threadIdx.x;
    const float* xb = x + (size_t)b * C_ * HW_;

    float acc[27];
#pragma unroll
    for (int j = 0; j < 27; j++) acc[j] = 0.f;

    for (int cc = 0; cc < 4; cc++) {           // 4 chunks of 16 channels
        __syncthreads();
        for (int i = tid; i < 16 * 18 * 18; i += 256) {
            int cl = i / 324, r = i % 324, ty = r / 18, tx = r % 18;
            int gh = h0 - 1 + ty, gw = w0 - 1 + tx;
            float v = 0.f;
            if ((unsigned)gh < (unsigned)H_ && (unsigned)gw < (unsigned)W_)
                v = xb[(size_t)(cc * 16 + cl) * HW_ + gh * W_ + gw];
            xt[cl * 360 + ty * 20 + tx] = v;
        }
        for (int i = tid; i < 27 * 144; i += 256) {
            int j = i / 144, r = i % 144;
            wl[j * 144 + r] = ow[(size_t)j * CK_ + cc * 144 + r];
        }
        __syncthreads();
        for (int c = 0; c < 16; c++) {
            float xv[9];
#pragma unroll
            for (int t = 0; t < 9; t++)
                xv[t] = xt[c * 360 + (threadIdx.y + t / 3) * 20 + threadIdx.x + t % 3];
#pragma unroll
            for (int j = 0; j < 27; j++) {
                float s = acc[j];
#pragma unroll
                for (int t = 0; t < 9; t++) s += xv[t] * wl[j * 144 + c * 9 + t];
                acc[j] = s;
            }
        }
    }
    size_t p = (size_t)b * HW_ + (size_t)(h0 + threadIdx.y) * W_ + (w0 + threadIdx.x);
    float* o = om + p * 27;
#pragma unroll
    for (int j = 0; j < 18; j++) o[j] = acc[j] + obv[j];
#pragma unroll
    for (int j = 18; j < 27; j++) {
        float v = acc[j] + obv[j];
        o[j] = 1.f / (1.f + __expf(-v));
    }
}

// ---------------------------------------------------------------------------
// Kernel 2: fused deformable sampling (bf16 A-tile in LDS) + WMMA GEMM.
// Per block: M=128 pixels, N=64 outputs, K=576 in 18 steps of wmma bf16 k32.
// ---------------------------------------------------------------------------
__global__ __launch_bounds__(256) void dcn_main(const float* __restrict__ x,
                                                const float* __restrict__ om,
                                                const unsigned short* __restrict__ wbf,
                                                const float* __restrict__ bias,
                                                float* __restrict__ out) {
    extern __shared__ __align__(16) char smem3[];
    unsigned short* A  = (unsigned short*)smem3;        // [MTILE][LDA_]
    unsigned short* Bw = A + MTILE * LDA_;              // [CO_][LDA_]

    const int tid = threadIdx.x;
    const int P0  = blockIdx.x * MTILE;
    const int b   = P0 / HW_;
    const int hw0 = P0 % HW_;

    // ---- Phase A: bf16 weights -> LDS (contiguous source, padded dest rows)
    {
        const uint4* src = (const uint4*)wbf;
        for (int v = tid; v < (CO_ * CK_) / 8; v += 256) {
            int o = v / 72, kq = v % 72;                 // 72 = 576/8
            *(uint4*)&Bw[o * LDA_ + kq * 8] = src[v];
        }
    }

    // ---- Phase B: bilinear deformable sampling -> bf16 A-tile in LDS
    const float* xb = x + (size_t)b * C_ * HW_;
    for (int task = tid; task < MTILE * KK_; task += 256) {
        int m  = task / KK_;
        int kk = task - m * KK_;
        int hw = hw0 + m;
        int hy = hw >> 7, wxp = hw & 127;
        const float* of = om + ((size_t)P0 + m) * 27;
        float gx  = (float)(wxp + (kk % 3) - 1) + of[kk];
        float gy  = (float)(hy  + (kk / 3) - 1) + of[9 + kk];
        float msk = of[18 + kk];
        float x0f = floorf(gx), y0f = floorf(gy);
        int xi0 = (int)x0f, yi0 = (int)y0f, xi1 = xi0 + 1, yi1 = yi0 + 1;
        float fx = gx - x0f, fy = gy - y0f;
        float vx0 = ((unsigned)xi0 < (unsigned)W_) ? 1.f : 0.f;
        float vx1 = ((unsigned)xi1 < (unsigned)W_) ? 1.f : 0.f;
        float vy0 = ((unsigned)yi0 < (unsigned)H_) ? 1.f : 0.f;
        float vy1 = ((unsigned)yi1 < (unsigned)H_) ? 1.f : 0.f;
        int xc0 = iclamp(xi0, 0, W_ - 1), xc1 = iclamp(xi1, 0, W_ - 1);
        int yc0 = iclamp(yi0, 0, H_ - 1), yc1 = iclamp(yi1, 0, H_ - 1);
        float w00 = (1.f - fy) * (1.f - fx) * vy0 * vx0 * msk;
        float w01 = (1.f - fy) * fx         * vy0 * vx1 * msk;
        float w10 = fy * (1.f - fx)         * vy1 * vx0 * msk;
        float w11 = fy * fx                 * vy1 * vx1 * msk;
        int o00 = yc0 * W_ + xc0, o01 = yc0 * W_ + xc1;
        int o10 = yc1 * W_ + xc0, o11 = yc1 * W_ + xc1;
        unsigned short* arow = A + m * LDA_ + kk;
        for (int c = 0; c < C_; c++) {
            const float* xc = xb + (size_t)c * HW_;
            float v = w00 * xc[o00] + w01 * xc[o01] + w10 * xc[o10] + w11 * xc[o11];
            arow[c * KK_] = f2bf(v);
        }
    }
    __syncthreads();

    // ---- Phase C: WMMA. Wave w: rows [16w, 16w+16), N = 0..63 (4 fragments).
    const int wave = tid >> 5;
    const int lane = tid & 31;
    const int lr   = lane & 15;
    const int kbA  = (lane < 16) ? 0 : 8;    // 16-bit A: K 0..7/16..23 vs 8..15/24..31
    const int kbB  = (lane < 16) ? 0 : 8;    // 16-bit B: K 0..7/16..23 vs 8..15/24..31

    v8f acc[4];
#pragma unroll
    for (int j = 0; j < 4; j++) {
        float bb = bias[j * 16 + lr];
        v8f a = {bb, bb, bb, bb, bb, bb, bb, bb};
        acc[j] = a;
    }

    const unsigned short* arow = A + (wave * 16 + lr) * LDA_;
    for (int kt = 0; kt < CK_ / 32; kt++) {
        FragBF af;
        const unsigned short* ap = arow + kt * 32 + kbA;
        af.u[0] = *(const uint4*)(ap);
        af.u[1] = *(const uint4*)(ap + 16);
#pragma unroll
        for (int j = 0; j < 4; j++) {
            FragBF bf;
            const unsigned short* bp = Bw + (j * 16 + lr) * LDA_ + kt * 32 + kbB;
            bf.u[0] = *(const uint4*)(bp);
            bf.u[1] = *(const uint4*)(bp + 16);
            acc[j] = __builtin_amdgcn_wmma_f32_16x16x32_bf16(
                false, af.v, false, bf.v, (short)0, acc[j], false, false);
        }
    }

    // ---- Phase D: store per 16x16 f32 C/D layout (+ bias already in acc)
    const int mrow0 = wave * 16 + ((lane < 16) ? 0 : 8);
    float* obase = out + (size_t)b * CO_ * HW_ + hw0;
#pragma unroll
    for (int j = 0; j < 4; j++) {
        int o = j * 16 + lr;
        float* op = obase + (size_t)o * HW_;
#pragma unroll
        for (int r = 0; r < 8; r++) {
            op[mrow0 + r] = acc[j][r];
        }
    }
}

// ---------------------------------------------------------------------------
extern "C" void kernel_launch(void* const* d_in, const int* in_sizes, int n_in,
                              void* d_out, int out_size, void* d_ws, size_t ws_size,
                              hipStream_t stream) {
    (void)in_sizes; (void)n_in; (void)out_size; (void)ws_size;
    const float* x   = (const float*)d_in[0];   // (8,64,128,128)
    const float* ow  = (const float*)d_in[1];   // (27,64,3,3)
    const float* obv = (const float*)d_in[2];   // (27,)
    const float* w   = (const float*)d_in[3];   // (64,64,3,3)
    const float* bs  = (const float*)d_in[4];   // (64,)
    float* out = (float*)d_out;                 // (8,64,128,128)

    float* ws_off = (float*)d_ws;                                   // [B*HW][27] f32
    unsigned short* ws_wbf =
        (unsigned short*)((char*)d_ws + (size_t)B_ * HW_ * 27 * sizeof(float));

    dcn_prep_w<<<(CO_ * CK_ + 255) / 256, 256, 0, stream>>>(w, ws_wbf);

    dim3 g2(W_ / 16, H_ / 16, B_), b2(16, 16);
    dcn_offsets<<<g2, b2, 0, stream>>>(x, ow, obv, ws_off);

    size_t lds = (size_t)(MTILE * LDA_ + CO_ * LDA_) * sizeof(unsigned short); // 224256 B
    dcn_main<<<(B_ * HW_) / MTILE, 256, lds, stream>>>(x, ws_off, ws_wbf, bs, out);
}
